// DynamicReasoner_52613349376828
// MI455X (gfx1250) — matrix operand order repaired and checked by reading, not compiled, not run on hardware
//
#include <hip/hip_runtime.h>
#include <hip/hip_bf16.h>

// ---------------------------------------------------------------------------
// Problem sizes (fixed by the reference): B=16, N=1024, D=512, HEAD=256
// ---------------------------------------------------------------------------
#define BB    16
#define NN    1024
#define DD    512
#define HH    256
#define MTOT  (BB * NN)          // 16384 rows total

typedef __bf16 bf16;
typedef __attribute__((ext_vector_type(16))) __bf16        v16bf;
typedef __attribute__((ext_vector_type(4)))  __bf16        v4bf;
typedef __attribute__((ext_vector_type(8)))  float         v8f;
typedef __attribute__((ext_vector_type(4)))  unsigned int  v4u;

union FragU { v4u q[2]; v16bf v; };

__device__ __forceinline__ float fast_rcp(float x) {
    return __builtin_amdgcn_rcpf(x);
}
// tanh(x) = 1 - 2/(exp(2x)+1); exp saturation gives correct +-1 tails.
__device__ __forceinline__ float fast_tanh(float x) {
    return 1.0f - 2.0f * fast_rcp(__expf(2.0f * x) + 1.0f);
}

// ---------------------------------------------------------------------------
// f32 -> bf16 convert, 4 elements/thread (n must be a multiple of 4)
// ---------------------------------------------------------------------------
__global__ void k_cvt(const float* __restrict__ src, bf16* __restrict__ dst, int n) {
    int i = (blockIdx.x * blockDim.x + threadIdx.x) * 4;
    if (i < n) {
        float4 v = *(const float4*)(src + i);
        v4bf o = {(bf16)v.x, (bf16)v.y, (bf16)v.z, (bf16)v.w};
        *(v4bf*)(dst + i) = o;
    }
}

// f32 [K x N] -> bf16 transposed [N x K]:  dst[n*K+k] = src[k*N+n]
__global__ void k_cvt_t(const float* __restrict__ src, bf16* __restrict__ dst,
                        int K, int N) {
    int idx = blockIdx.x * blockDim.x + threadIdx.x;
    if (idx < K * N) {
        int k = idx / N, n = idx - k * N;
        dst[(size_t)n * K + k] = (bf16)src[idx];
    }
}

// ---------------------------------------------------------------------------
// Tiled bf16 WMMA GEMM:  out = epilogue(A[M x K] @ B[K x N] + bias)
// B is passed PRE-TRANSPOSED as Bt[N x K] so both LDS tiles fill with b128 ops.
// Block tile 128x128, 8 waves of 32 lanes, wave tile 32x64 (2x4 WMMA frags).
// MODE 0: f32 out = acc (+bias)
// MODE 1: bf16 out = tanh(acc + bias)
// MODE 2: bf16 out = relu((acc + 2*bias) * rdenom[row]), written at colOff
// MODE 3: f32 out = acc + bias  (final projection)
// Requires M%128==0, N%128==0, K%32==0 (true for all calls here).
// ---------------------------------------------------------------------------
template <int MODE>
__global__ __launch_bounds__(256)
void k_gemm(const bf16* __restrict__ A, int lda,
            const bf16* __restrict__ Bt, int ldb,   // ldb == K
            const float* __restrict__ bias,
            const float* __restrict__ rdenom,
            void* __restrict__ out, int ldo, int colOff, int K) {
    __shared__ bf16 As[128 * 40];   // A tile rows (M x K-slab), row stride 40
    __shared__ bf16 Bs[128 * 40];   // Bt tile rows (N x K-slab), row stride 40

    const int t    = threadIdx.x;
    const int lane = t & 31;
    const int wave = t >> 5;
    const int wm   = wave >> 1;     // 0..3  -> 32-row slabs
    const int wn   = wave & 1;      // 0..1  -> 64-col slabs
    const int gm   = blockIdx.y * 128;
    const int gn   = blockIdx.x * 128;

    v8f acc[2][4];
#pragma unroll
    for (int r = 0; r < 2; ++r)
#pragma unroll
        for (int c = 0; c < 4; ++c)
#pragma unroll
            for (int e = 0; e < 8; ++e) acc[r][c][e] = 0.0f;

    const int klA = (lane & 16) ? 8 : 0;    // A layout: half-wave K offset
    const int klB = (lane & 16) ? 8 : 0;    // B layout: half-wave K offset
    const int row = t >> 1, col = (t & 1) * 16;   // tile-fill coords

    for (int k0 = 0; k0 < K; k0 += 32) {
        __syncthreads();
        {   // A tile fill: 128x32, 16 bf16 per thread (2x b128)
            const v4u* srcA = (const v4u*)(A + (size_t)(gm + row) * lda + k0 + col);
            v4u a0 = srcA[0], a1 = srcA[1];
            *(v4u*)(&As[row * 40 + col])     = a0;
            *(v4u*)(&As[row * 40 + col + 8]) = a1;
        }
        {   // B tile fill from Bt: rows are N, cols are K -> same pattern
            const v4u* srcB = (const v4u*)(Bt + (size_t)(gn + row) * ldb + k0 + col);
            v4u b0 = srcB[0], b1 = srcB[1];
            *(v4u*)(&Bs[row * 40 + col])     = b0;
            *(v4u*)(&Bs[row * 40 + col + 8]) = b1;
        }
        __syncthreads();

        FragU afr[2], bfr[4];
#pragma unroll
        for (int r = 0; r < 2; ++r) {
            int m = wm * 32 + r * 16 + (lane & 15);
            afr[r].q[0] = *(const v4u*)(&As[m * 40 + klA]);
            afr[r].q[1] = *(const v4u*)(&As[m * 40 + 16 + klA]);
        }
#pragma unroll
        for (int c = 0; c < 4; ++c) {
            int n = wn * 64 + c * 16 + (lane & 15);
            bfr[c].q[0] = *(const v4u*)(&Bs[n * 40 + klB]);
            bfr[c].q[1] = *(const v4u*)(&Bs[n * 40 + klB + 16]);
        }
#pragma unroll
        for (int r = 0; r < 2; ++r)
#pragma unroll
            for (int c = 0; c < 4; ++c)
                acc[r][c] = __builtin_amdgcn_wmma_f32_16x16x32_bf16(
                    false, afr[r].v, false, bfr[c].v,
                    (short)0, acc[r][c], false, false);
    }

    // Epilogue. C/D layout: lanes 0-15 N=lane M=vgpr ; lanes 16-31 N=lane-16 M=vgpr+8
#pragma unroll
    for (int r = 0; r < 2; ++r)
#pragma unroll
        for (int c = 0; c < 4; ++c) {
            int ocol = gn + wn * 64 + c * 16 + (lane & 15);
            float bv = bias ? bias[ocol] : 0.0f;
#pragma unroll
            for (int e = 0; e < 8; ++e) {
                int orow = gm + wm * 32 + r * 16 + e + ((lane & 16) ? 8 : 0);
                float v = acc[r][c][e];
                if (MODE == 0) {
                    ((float*)out)[(size_t)orow * ldo + ocol] = v + bv;
                } else if (MODE == 1) {
                    ((bf16*)out)[(size_t)orow * ldo + ocol] =
                        (bf16)fast_tanh(v + bv);
                } else if (MODE == 2) {
                    float g = (v + 2.0f * bv) * rdenom[orow];
                    ((bf16*)out)[(size_t)orow * ldo + colOff + ocol] =
                        (bf16)fmaxf(g, 0.0f);
                } else {
                    ((float*)out)[(size_t)orow * ldo + ocol] = v + bv;
                }
            }
        }
}

// ---------------------------------------------------------------------------
// f[row] = exp( x[row,:] . fi_w )   (one wave per row)
// ---------------------------------------------------------------------------
__global__ void k_fvec(const float* __restrict__ x, const float* __restrict__ fw,
                       float* __restrict__ f) {
    int row  = blockIdx.x * 8 + (threadIdx.x >> 5);
    int lane = threadIdx.x & 31;
    const float* xr = x + (size_t)row * DD;
    float s = 0.0f;
    for (int c = lane; c < DD; c += 32) s += xr[c] * fw[c];
#pragma unroll
    for (int off = 16; off > 0; off >>= 1) s += __shfl_xor(s, off, 32);
    if (lane == 0) f[row] = __expf(s);
}

// ---------------------------------------------------------------------------
// e[row] = exp( sum_c u[row,c] * tc[row,c] )   (one wave per row)
// ---------------------------------------------------------------------------
__global__ void k_se(const float* __restrict__ u, const bf16* __restrict__ tc,
                     float* __restrict__ e) {
    int row  = blockIdx.x * 8 + (threadIdx.x >> 5);
    int lane = threadIdx.x & 31;
    const float* ur = u  + (size_t)row * DD;
    const bf16* tr  = tc + (size_t)row * DD;
    float s = 0.0f;
    for (int c = lane; c < DD; c += 32) s += ur[c] * (float)tr[c];
#pragma unroll
    for (int off = 16; off > 0; off >>= 1) s += __shfl_xor(s, off, 32);
    if (lane == 0) e[row] = __expf(s);
}

// ---------------------------------------------------------------------------
// Per-batch scalars: sc[b] = {S, F, 1/S, 1/(S*F)}
// ---------------------------------------------------------------------------
__global__ void k_bscal(const float* __restrict__ e, const float* __restrict__ f,
                        float* __restrict__ sc) {
    int b = blockIdx.x, t = threadIdx.x;
    float S = 0.0f, F = 0.0f;
    for (int j = t; j < NN; j += 256) {
        float ev = e[b * NN + j];
        S += ev;
        F += ev * f[b * NN + j];
    }
    __shared__ float sS[256], sF[256];
    sS[t] = S; sF[t] = F;
    __syncthreads();
    for (int off = 128; off > 0; off >>= 1) {
        if (t < off) { sS[t] += sS[t + off]; sF[t] += sF[t + off]; }
        __syncthreads();
    }
    if (t == 0) {
        float Sv = sS[0], Fv = sF[0];
        sc[b * 4 + 0] = Sv;
        sc[b * 4 + 1] = Fv;
        sc[b * 4 + 2] = 1.0f / Sv;
        sc[b * 4 + 3] = 1.0f / (Sv * Fv);
    }
}

// rden[row] = 1 / (2 - e*f/F_b)   (reciprocal so GEMM epilogue multiplies)
__global__ void k_denom(const float* __restrict__ e, const float* __restrict__ f,
                        const float* __restrict__ sc, float* __restrict__ rden) {
    int row = blockIdx.x * blockDim.x + threadIdx.x;
    if (row < MTOT) {
        int b = row >> 10;
        rden[row] = 1.0f / (2.0f - e[row] * f[row] / sc[b * 4 + 1]);
    }
}

// ---------------------------------------------------------------------------
// Per-batch column reductions over H = [H0 | H1] (split at `split` columns):
//   p[b,c] = sum_j e_j H[j,c],  q[b,c] = sum_j e_j f_j H[j,c]
// Grid: (C/256, BB). e,f staged in LDS.
// ---------------------------------------------------------------------------
__global__ void k_colred(const bf16* __restrict__ H0, int ld0, int split,
                         const bf16* __restrict__ H1, int ld1,
                         const float* __restrict__ e, const float* __restrict__ f,
                         float* __restrict__ p, float* __restrict__ q, int C) {
    int b = blockIdx.y;
    int c = blockIdx.x * 256 + threadIdx.x;
    __shared__ float se[NN], sf[NN];
    for (int j = threadIdx.x; j < NN; j += 256) {
        se[j] = e[b * NN + j];
        sf[j] = f[b * NN + j];
    }
    __syncthreads();
    float P = 0.0f, Q = 0.0f;
    if (c < split) {
        for (int j = 0; j < NN; ++j) {
            float h = (float)H0[(size_t)(b * NN + j) * ld0 + c];
            float w = se[j];
            P += w * h; Q += w * sf[j] * h;
        }
    } else {
        for (int j = 0; j < NN; ++j) {
            float h = (float)H1[(size_t)(b * NN + j) * ld1 + (c - split)];
            float w = se[j];
            P += w * h; Q += w * sf[j] * h;
        }
    }
    p[b * C + c] = P;
    q[b * C + c] = Q;
}

// ---------------------------------------------------------------------------
// Z[j,c] = p/S + (e_j/(SF)) q - (e_j f_j/(SF)) p + (1 - e_j/S) H[j,c]
// (this is (adj @ H + H)[j,c] via the rank-3 + diagonal form of adj)
// ---------------------------------------------------------------------------
__global__ void k_buildZ(const bf16* __restrict__ H0, int ld0, int split,
                         const bf16* __restrict__ H1, int ld1,
                         const float* __restrict__ e, const float* __restrict__ f,
                         const float* __restrict__ p, const float* __restrict__ q,
                         const float* __restrict__ sc,
                         bf16* __restrict__ Z, int C) {
    int idx = blockIdx.x * 256 + threadIdx.x;
    if (idx >= MTOT * C) return;
    int c   = idx % C;
    int row = idx / C;
    int b   = row >> 10;
    float invS  = sc[b * 4 + 2];
    float invSF = sc[b * 4 + 3];
    float ev = e[row], fv = f[row];
    float h  = (c < split) ? (float)H0[(size_t)row * ld0 + c]
                           : (float)H1[(size_t)row * ld1 + (c - split)];
    float pc = p[b * C + c], qc = q[b * C + c];
    float z = pc * invS + ev * invSF * qc - ev * fv * invSF * pc
            + (1.0f - ev * invS) * h;
    Z[(size_t)row * C + c] = (bf16)z;
}

// gcat = concat(g0,g1) + x   (bf16 out)
__global__ void k_gcat(const bf16* __restrict__ g, const float* __restrict__ x,
                       bf16* __restrict__ out) {
    int i = blockIdx.x * 256 + threadIdx.x;
    if (i < MTOT * DD) out[i] = (bf16)((float)g[i] + x[i]);
}

// ---------------------------------------------------------------------------
// Workspace layout (bytes). Transposed bf16 weights: Wt[n*K+k].
// ---------------------------------------------------------------------------
#define O_XBF   ((size_t)0)                        // 16384*512 bf16 = 16 MB
#define O_TPW   (O_XBF  + (size_t)MTOT * DD * 2)
#define O_TCW   (O_TPW  + (size_t)DD * DD * 2)
#define O_BILW  (O_TCW  + (size_t)DD * DD * 2)
#define O_W0    (O_BILW + (size_t)DD * DD * 2)
#define O_W1    (O_W0   + (size_t)DD * HH * 2)
#define O_OUTW  (O_W1   + (size_t)(DD + HH) * HH * 2)
#define O_TP    (O_OUTW + (size_t)DD * DD * 2)     // tp bf16; later reused as gcat
#define O_TC    (O_TP   + (size_t)MTOT * DD * 2)
#define O_U     (O_TC   + (size_t)MTOT * DD * 2)   // u f32 (32MB); reused as Z bf16
#define O_G     (O_U    + (size_t)MTOT * DD * 4)   // [g0 | g1] bf16, ld 512
#define O_F     (O_G    + (size_t)MTOT * DD * 2)
#define O_E     (O_F    + (size_t)MTOT * 4)
#define O_DEN   (O_E    + (size_t)MTOT * 4)
#define O_SC    (O_DEN  + (size_t)MTOT * 4)
#define O_P     (O_SC   + (size_t)1024)
#define O_Q     (O_P    + (size_t)BB * (DD + HH) * 4)

extern "C" void kernel_launch(void* const* d_in, const int* in_sizes, int n_in,
                              void* d_out, int out_size, void* d_ws, size_t ws_size,
                              hipStream_t stream) {
    const float* x     = (const float*)d_in[0];
    const float* tp_w  = (const float*)d_in[1];
    const float* tp_b  = (const float*)d_in[2];
    const float* tc_w  = (const float*)d_in[3];
    const float* tc_b  = (const float*)d_in[4];
    const float* fi_w  = (const float*)d_in[5];
    const float* bil_w = (const float*)d_in[6];
    const float* w0    = (const float*)d_in[7];
    const float* b0    = (const float*)d_in[8];
    const float* w1    = (const float*)d_in[9];
    const float* b1    = (const float*)d_in[10];
    const float* out_w = (const float*)d_in[11];
    const float* out_b = (const float*)d_in[12];

    char* ws = (char*)d_ws;
    bf16* XBF  = (bf16*)(ws + O_XBF);
    bf16* TPW  = (bf16*)(ws + O_TPW);    // transposed 512x512
    bf16* TCW  = (bf16*)(ws + O_TCW);    // transposed 512x512
    bf16* BILW = (bf16*)(ws + O_BILW);   // transposed 512x512
    bf16* W0   = (bf16*)(ws + O_W0);     // transposed 256x512
    bf16* W1   = (bf16*)(ws + O_W1);     // transposed 256x768
    bf16* OUTW = (bf16*)(ws + O_OUTW);   // transposed 512x512
    bf16* TP   = (bf16*)(ws + O_TP);
    bf16* TC   = (bf16*)(ws + O_TC);
    float* U   = (float*)(ws + O_U);
    bf16* Z    = (bf16*)(ws + O_U);      // reuse (u dead after k_se)
    bf16* G    = (bf16*)(ws + O_G);
    float* F   = (float*)(ws + O_F);
    float* E   = (float*)(ws + O_E);
    float* RDEN= (float*)(ws + O_DEN);
    float* SC  = (float*)(ws + O_SC);
    float* P   = (float*)(ws + O_P);
    float* Q   = (float*)(ws + O_Q);
    bf16* GCAT = TP;                     // reuse (tp dead after u = tp@bil_w)

    const int C1 = DD + HH;              // 768

    // --- bf16 conversions (weights transposed for b128 GEMM tile fills) ---
    k_cvt<<<(MTOT * DD / 4 + 255) / 256, 256, 0, stream>>>(x, XBF, MTOT * DD);
    k_cvt_t<<<(DD * DD + 255) / 256, 256, 0, stream>>>(tp_w, TPW, DD, DD);
    k_cvt_t<<<(DD * DD + 255) / 256, 256, 0, stream>>>(tc_w, TCW, DD, DD);
    k_cvt_t<<<(DD * DD + 255) / 256, 256, 0, stream>>>(bil_w, BILW, DD, DD);
    k_cvt_t<<<(DD * HH + 255) / 256, 256, 0, stream>>>(w0, W0, DD, HH);
    k_cvt_t<<<(C1 * HH + 255) / 256, 256, 0, stream>>>(w1, W1, C1, HH);
    k_cvt_t<<<(DD * DD + 255) / 256, 256, 0, stream>>>(out_w, OUTW, DD, DD);

    // --- tp = tanh(x@tp_w+b), tc = tanh(x@tc_w+b), u = tp@bil_w -----------
    dim3 gD(DD / 128, MTOT / 128);       // 4 x 128
    k_gemm<1><<<gD, 256, 0, stream>>>(XBF, DD, TPW, DD, tp_b, nullptr, TP, DD, 0, DD);
    k_gemm<1><<<gD, 256, 0, stream>>>(XBF, DD, TCW, DD, tc_b, nullptr, TC, DD, 0, DD);
    k_gemm<0><<<gD, 256, 0, stream>>>(TP, DD, BILW, DD, nullptr, nullptr, U, DD, 0, DD);

    // --- f, e (=exp(s)), per-batch scalars, reciprocal denom --------------
    k_fvec<<<MTOT / 8, 256, 0, stream>>>(x, fi_w, F);
    k_se<<<MTOT / 8, 256, 0, stream>>>(U, TC, E);
    k_bscal<<<BB, 256, 0, stream>>>(E, F, SC);
    k_denom<<<(MTOT + 255) / 256, 256, 0, stream>>>(E, F, SC, RDEN);

    // --- GCN layer 0: H = x ----------------------------------------------
    k_colred<<<dim3(DD / 256, BB), 256, 0, stream>>>(XBF, DD, DD, nullptr, 0,
                                                     E, F, P, Q, DD);
    k_buildZ<<<(MTOT * DD + 255) / 256, 256, 0, stream>>>(
        XBF, DD, DD, nullptr, 0, E, F, P, Q, SC, Z, DD);
    k_gemm<2><<<dim3(HH / 128, MTOT / 128), 256, 0, stream>>>(
        Z, DD, W0, DD, b0, RDEN, G, DD, 0, DD);

    // --- GCN layer 1: H = [x | g0] ---------------------------------------
    k_colred<<<dim3(C1 / 256, BB), 256, 0, stream>>>(XBF, DD, DD, G, DD,
                                                     E, F, P, Q, C1);
    k_buildZ<<<(MTOT * C1 + 255) / 256, 256, 0, stream>>>(
        XBF, DD, DD, G, DD, E, F, P, Q, SC, Z, C1);
    k_gemm<2><<<dim3(HH / 128, MTOT / 128), 256, 0, stream>>>(
        Z, C1, W1, C1, b1, RDEN, G, DD, HH, C1);

    // --- gcn_out = [g0|g1] + x ; final projection -------------------------
    k_gcat<<<(MTOT * DD + 255) / 256, 256, 0, stream>>>(G, x, GCAT);
    k_gemm<3><<<gD, 256, 0, stream>>>(GCAT, DD, OUTW, DD, out_b, nullptr,
                                      (float*)d_out, DD, 0, DD);
}
